// UnifiedMSA_77850577207523
// MI455X (gfx1250) — matrix operand order repaired and checked by reading, not compiled, not run on hardware
//
#include <hip/hip_runtime.h>
#include <hip/hip_bf16.h>

// ---------------- constants (match reference) ----------------
#define BATCH 2
#define NSEQ  1024
#define DIM   1024
#define NH    16
#define DKH   64
#define BH    (BATCH * NH)
#define SCALE 0.125f          // 1/sqrt(64)
#define T1C   0.1f
#define T2C   0.1f

typedef __bf16 bf16_t;
typedef __attribute__((ext_vector_type(16))) __bf16 v16bf;
typedef __attribute__((ext_vector_type(8)))  __bf16 v8bf;
typedef __attribute__((ext_vector_type(8)))  float  v8f;

#define NEG_INF (-__builtin_inff())

__device__ __forceinline__ v8f wmma_bf16(v16bf a, v16bf b, v8f c) {
  // D = A(16x32 bf16) x B(32x16 bf16) + C(16x16 f32)
  return __builtin_amdgcn_wmma_f32_16x16x32_bf16(false, a, false, b, (short)0, c,
                                                 false, false);
}

__device__ __forceinline__ v8f v8f_zero() {
  v8f z = {0.f, 0.f, 0.f, 0.f, 0.f, 0.f, 0.f, 0.f};
  return z;
}

// A-operand tile (16 rows x 32 K, bf16), in this file's k order (matches load_cols):
// lane L holds row L&15; lanes 0-15: K 0..15; lanes 16-31: K 16..31 (contiguous).
__device__ __forceinline__ v16bf load_rows(const bf16_t* p0, int ld) {
  int lane = threadIdx.x & 31;
  const bf16_t* p = p0 + (lane & 15) * ld + ((lane >> 4) << 4);
  v8bf lo = *(const v8bf*)p;          // K base .. +7
  v8bf hi = *(const v8bf*)(p + 8);    // K base+8 .. +15
  v16bf r;
#pragma unroll
  for (int i = 0; i < 8; ++i) { r[i] = lo[i]; r[i + 8] = hi[i]; }
  return r;
}

// B-operand tile (32 K x 16 cols, bf16): lane L holds column L&15,
// K 0..15 (lanes 0-15) or K 16..31 (lanes 16-31), contiguous.
// Columns of B are rows of the source array (since B = W^T / K^T).
__device__ __forceinline__ v16bf load_cols(const bf16_t* p0, int ld) {
  int lane = threadIdx.x & 31;
  const bf16_t* p = p0 + (lane & 15) * ld + ((lane >> 4) << 4);
  v8bf lo = *(const v8bf*)p;
  v8bf hi = *(const v8bf*)(p + 8);
  v16bf r;
#pragma unroll
  for (int i = 0; i < 8; ++i) { r[i] = lo[i]; r[i + 8] = hi[i]; }
  return r;
}

// Row reduction across the 16 lanes holding one C/D row (wave32 half-wave).
__device__ __forceinline__ float halfmax(float v) {
#pragma unroll
  for (int m = 1; m <= 8; m <<= 1) v = fmaxf(v, __shfl_xor(v, m, 32));
  return v;
}
__device__ __forceinline__ float halfsum(float v) {
#pragma unroll
  for (int m = 1; m <= 8; m <<= 1) v += __shfl_xor(v, m, 32);
  return v;
}

// online softmax (max, sum-exp) update for one row with a 16-col tile slice
__device__ __forceinline__ void online_upd(float& m, float& z, float s) {
  float tm   = halfmax(s);
  float mnew = fmaxf(m, tm);
  float e    = (s == NEG_INF) ? 0.f : __expf(s - mnew);
  float es   = halfsum(e);
  float r    = (m == mnew) ? 1.f : __expf(m - mnew);
  z = z * r + es;
  m = mnew;
}

struct Six { v8f p11, p12, p21, p22, t1, t2; };
__device__ __forceinline__ Six six_zero() {
  Six s;
  s.p11 = v8f_zero(); s.p12 = v8f_zero(); s.p21 = v8f_zero();
  s.p22 = v8f_zero(); s.t1 = v8f_zero(); s.t2 = v8f_zero();
  return s;
}

// Accumulate the six 16x16 (K=64) score products for one column tile:
// p11=q1n.k1m  p12=q1n.k2m  p21=q2n.k1m  p22=q2n.k2m  t1=k1n.q1m  t2=k2n.q2m
__device__ __forceinline__ void accum_six(Six& S,
                                          const v16bf* Aq1, const v16bf* Aq2,
                                          const v16bf* Ak1, const v16bf* Ak2,
                                          const bf16_t* k1c, const bf16_t* k2c,
                                          const bf16_t* q1c, const bf16_t* q2c) {
  v16bf b0 = load_cols(k1c, DKH), b1 = load_cols(k1c + 32, DKH);
  S.p11 = wmma_bf16(Aq1[0], b0, S.p11); S.p11 = wmma_bf16(Aq1[1], b1, S.p11);
  S.p21 = wmma_bf16(Aq2[0], b0, S.p21); S.p21 = wmma_bf16(Aq2[1], b1, S.p21);
  b0 = load_cols(k2c, DKH); b1 = load_cols(k2c + 32, DKH);
  S.p12 = wmma_bf16(Aq1[0], b0, S.p12); S.p12 = wmma_bf16(Aq1[1], b1, S.p12);
  S.p22 = wmma_bf16(Aq2[0], b0, S.p22); S.p22 = wmma_bf16(Aq2[1], b1, S.p22);
  b0 = load_cols(q1c, DKH); b1 = load_cols(q1c + 32, DKH);
  S.t1 = wmma_bf16(Ak1[0], b0, S.t1);  S.t1 = wmma_bf16(Ak1[1], b1, S.t1);
  b0 = load_cols(q2c, DKH); b1 = load_cols(q2c + 32, DKH);
  S.t2 = wmma_bf16(Ak2[0], b0, S.t2);  S.t2 = wmma_bf16(Ak2[1], b1, S.t2);
}

// ================= K0: fp32 -> bf16 convert =================
__global__ void k_cvt(const float* __restrict__ src, bf16_t* __restrict__ dst, int n) {
  int i = blockIdx.x * blockDim.x + threadIdx.x;
  int stride = gridDim.x * blockDim.x;
  for (; i < n; i += stride) dst[i] = (bf16_t)src[i];
}

// ================= K1: fused QKV GEMMs (both weight sets) =================
// one wave computes a 16x64 output strip of x @ W^T (K=1024):
// A tile loaded once per k-step, reused across 4 B tiles.
// Single-basic-block inner loop: unconditional speculative prefetch, named
// accumulators (no register rotation), loads grouped ahead of the 4 wmmas.
__global__ void k_qkv(const bf16_t* __restrict__ xb,
                      const bf16_t* __restrict__ w1b, const bf16_t* __restrict__ w2b,
                      bf16_t* __restrict__ q1, bf16_t* __restrict__ k1,
                      bf16_t* __restrict__ v1t,
                      bf16_t* __restrict__ q2, bf16_t* __restrict__ k2) {
  int wid = blockIdx.x * (blockDim.x >> 5) + (threadIdx.x >> 5);
  const int TILES = 128 * 48;               // rowblocks x 64-col strips per set
  int s = wid / TILES;
  if (s >= 2) return;
  int rem = wid - s * TILES;
  int rt = rem / 48, ct = rem - rt * 48;
  int r0 = rt << 4, o0 = ct << 6;
  int part = o0 >> 10;                       // 0=q 1=k 2=v
  if (s == 1 && part == 2) return;           // v2 is unused by the reference

  const bf16_t* Wb  = s ? w2b : w1b;
  const bf16_t* arow = xb + r0 * DIM;
  const bf16_t* brow = Wb + o0 * DIM;
  v8f acc0 = v8f_zero(), acc1 = v8f_zero(), acc2 = v8f_zero(), acc3 = v8f_zero();
  for (int kk = 0; kk < 32; ++kk) {
    // speculative prefetch of next k-step (safe past the end; WGP scope)
    __builtin_prefetch(arow + ((kk + 1) << 5), 0, 3);
    v16bf a  = load_rows(arow + (kk << 5), DIM);
    v16bf b0 = load_cols(brow + (kk << 5), DIM);
    v16bf b1 = load_cols(brow + 16 * DIM + (kk << 5), DIM);
    v16bf b2 = load_cols(brow + 32 * DIM + (kk << 5), DIM);
    v16bf b3 = load_cols(brow + 48 * DIM + (kk << 5), DIM);
    acc0 = wmma_bf16(a, b0, acc0);
    acc1 = wmma_bf16(a, b1, acc1);
    acc2 = wmma_bf16(a, b2, acc2);
    acc3 = wmma_bf16(a, b3, acc3);
  }
  int lane = threadIdx.x & 31, c = lane & 15, half8 = (lane >> 4) << 3;
  int hh = (o0 >> 6) & 15;                   // head (uniform over the strip)
  v8f accs[4] = {acc0, acc1, acc2, acc3};
#pragma unroll
  for (int cc = 0; cc < 4; ++cc) {
    int dk = (cc << 4) + c;
#pragma unroll
    for (int i = 0; i < 8; ++i) {
      int r = r0 + i + half8;
      int b = r >> 10, n = r & 1023;
      int bh = b * NH + hh;
      bf16_t v = (bf16_t)accs[cc][i];
      if (s == 0) {
        if (part == 0)      q1[(bh * NSEQ + n) * DKH + dk] = v;
        else if (part == 1) k1[(bh * NSEQ + n) * DKH + dk] = v;
        else                v1t[(bh * DKH + dk) * NSEQ + n] = v;  // V stored transposed
      } else {
        if (part == 0)      q2[(bh * NSEQ + n) * DKH + dk] = v;
        else                k2[(bh * NSEQ + n) * DKH + dk] = v;
      }
    }
  }
}

// ================= K2: online softmax stats for S, S1, S2 =================
// one wave per (bh, 16-row block); sweeps all 64 column tiles
__global__ void k_attn_stats(const bf16_t* __restrict__ q1, const bf16_t* __restrict__ k1,
                             const bf16_t* __restrict__ q2, const bf16_t* __restrict__ k2,
                             const unsigned char* __restrict__ mask,
                             const float* __restrict__ mixp,
                             float* __restrict__ msS, float* __restrict__ zS,
                             float* __restrict__ ms1, float* __restrict__ z1,
                             float* __restrict__ ms2, float* __restrict__ z2) {
  int wid = blockIdx.x * (blockDim.x >> 5) + (threadIdx.x >> 5);
  int bh = wid >> 6, rb = wid & 63;
  if (bh >= BH) return;
  int r0 = rb << 4;
  int lane = threadIdx.x & 31, half8 = (lane >> 4) << 3;
  int b = bh >> 4;
  float mix00 = mixp[0], mix01 = mixp[1], mix10 = mixp[2], mix11 = mixp[3];

  v16bf Aq1[2], Aq2[2], Ak1[2], Ak2[2];
#pragma unroll
  for (int ks = 0; ks < 2; ++ks) {
    int off = (bh * NSEQ + r0) * DKH + (ks << 5);
    Aq1[ks] = load_rows(q1 + off, DKH);
    Aq2[ks] = load_rows(q2 + off, DKH);
    Ak1[ks] = load_rows(k1 + off, DKH);
    Ak2[ks] = load_rows(k2 + off, DKH);
  }

  float mS_[8], zS_[8], m1_[8], z1_[8], m2_[8], z2_[8];
#pragma unroll
  for (int i = 0; i < 8; ++i) {
    mS_[i] = NEG_INF; m1_[i] = NEG_INF; m2_[i] = NEG_INF;
    zS_[i] = 0.f; z1_[i] = 0.f; z2_[i] = 0.f;
  }

  for (int mt = 0; mt < 64; ++mt) {
    int m0 = mt << 4;
    int cb = (bh * NSEQ + m0) * DKH;
    Six S = six_zero();
    accum_six(S, Aq1, Aq2, Ak1, Ak2, k1 + cb, k2 + cb, q1 + cb, q2 + cb);
    int col = m0 + (lane & 15);
#pragma unroll
    for (int i = 0; i < 8; ++i) {
      int row = r0 + i + half8;
      bool valid = mask[(b * NSEQ + row) * NSEQ + col] != 0;
      float s1v = SCALE * S.p11[i];
      float s2v = SCALE * S.p22[i];
      float sm  = SCALE * (mix00 * S.p11[i] + mix01 * S.p12[i] +
                           mix10 * S.p21[i] + mix11 * S.p22[i] +
                           T1C * S.t1[i] + T2C * S.t2[i]);
      if (!valid) { s1v = NEG_INF; s2v = NEG_INF; sm = NEG_INF; }
      online_upd(mS_[i], zS_[i], sm);
      online_upd(m1_[i], z1_[i], s1v);
      online_upd(m2_[i], z2_[i], s2v);
    }
  }
  if ((lane & 15) == 0) {
#pragma unroll
    for (int i = 0; i < 8; ++i) {
      int idx = bh * NSEQ + r0 + i + half8;
      msS[idx] = mS_[i]; zS[idx] = zS_[i];
      ms1[idx] = m1_[i]; z1[idx] = z1_[i];
      ms2[idx] = m2_[i]; z2[idx] = z2_[i];
    }
  }
}

// ================= K3: A2 row sums + argmax (anchor index) =================
__global__ void k_rowsum_argmax(const bf16_t* __restrict__ q2, const bf16_t* __restrict__ k2,
                                const unsigned char* __restrict__ mask,
                                const float* __restrict__ ms2, const float* __restrict__ z2,
                                int* __restrict__ kstar) {
  __shared__ float rs[NSEQ];
  int bh = blockIdx.x, b = bh >> 4;
  int w = threadIdx.x >> 5;
  int lane = threadIdx.x & 31, half8 = (lane >> 4) << 3;

  for (int rb = w; rb < 64; rb += 8) {
    int r0 = rb << 4;
    v16bf Aq2[2];
    Aq2[0] = load_rows(q2 + (bh * NSEQ + r0) * DKH, DKH);
    Aq2[1] = load_rows(q2 + (bh * NSEQ + r0) * DKH + 32, DKH);
    float m2_[8], z2i_[8], acc_[8];
#pragma unroll
    for (int i = 0; i < 8; ++i) {
      int idx = bh * NSEQ + r0 + i + half8;
      m2_[i] = ms2[idx]; z2i_[i] = 1.f / z2[idx]; acc_[i] = 0.f;
    }
    for (int mt = 0; mt < 64; ++mt) {
      int m0 = mt << 4;
      const bf16_t* kc = k2 + (bh * NSEQ + m0) * DKH;
      v8f p = v8f_zero();
      v16bf b0 = load_cols(kc, DKH), b1 = load_cols(kc + 32, DKH);
      p = wmma_bf16(Aq2[0], b0, p);
      p = wmma_bf16(Aq2[1], b1, p);
      int col = m0 + (lane & 15);
#pragma unroll
      for (int i = 0; i < 8; ++i) {
        int row = r0 + i + half8;
        bool valid = mask[(b * NSEQ + row) * NSEQ + col] != 0;
        acc_[i] += valid ? __expf(SCALE * p[i] - m2_[i]) * z2i_[i] : 0.f;
      }
    }
#pragma unroll
    for (int i = 0; i < 8; ++i) {
      float t = halfsum(acc_[i]);
      if ((lane & 15) == 0) rs[r0 + i + half8] = t;
    }
  }
  __syncthreads();
  if (threadIdx.x == 0) {          // strict > : first-index tie-break, like argmax
    float best = rs[0]; int bi = 0;
    for (int n = 1; n < NSEQ; ++n) { float v = rs[n]; if (v > best) { best = v; bi = n; } }
    kstar[bh] = bi;
  }
}

// ================= K4: anchor row A2[k*, :] =================
__global__ void k_anchor(const bf16_t* __restrict__ q2, const bf16_t* __restrict__ k2,
                         const unsigned char* __restrict__ mask,
                         const float* __restrict__ ms2, const float* __restrict__ z2,
                         const int* __restrict__ kstar, float* __restrict__ anchor) {
  int bh = blockIdx.x, b = bh >> 4;
  int ks = kstar[bh];
  float m  = ms2[bh * NSEQ + ks];
  float zi = 1.f / z2[bh * NSEQ + ks];
  const bf16_t* qrow = q2 + (bh * NSEQ + ks) * DKH;
  for (int mc = threadIdx.x; mc < NSEQ; mc += blockDim.x) {
    const bf16_t* krow = k2 + (bh * NSEQ + mc) * DKH;
    float d = 0.f;
#pragma unroll 8
    for (int t = 0; t < DKH; ++t) d += (float)qrow[t] * (float)krow[t];
    bool valid = mask[(b * NSEQ + ks) * NSEQ + mc] != 0;
    anchor[bh * NSEQ + mc] = valid ? __expf(SCALE * d - m) * zi : 0.f;
  }
}

// ================= K5: fused final attention (recompute + P.V) =================
// one wave per (bh, 16-row block). Recomputes score tiles, forms
// P_mix = A_mix and P_sharp = A1 * anchor[m] in fp32, stages them in LDS as
// bf16 16x32 A-operands, then accumulates Y_mix, U (unnormalized sharp) and
// alpha (sharp row-sum via ones-column B) with WMMA.
__global__ void k_attn_out(const bf16_t* __restrict__ q1, const bf16_t* __restrict__ k1,
                           const bf16_t* __restrict__ q2, const bf16_t* __restrict__ k2,
                           const bf16_t* __restrict__ v1t,
                           const unsigned char* __restrict__ mask,
                           const float* __restrict__ mixp,
                           const float* __restrict__ msS, const float* __restrict__ zS,
                           const float* __restrict__ ms1, const float* __restrict__ z1,
                           const float* __restrict__ anchor,
                           bf16_t* __restrict__ Yb) {
  __shared__ __align__(32) bf16_t smPM[4][16 * 32];
  __shared__ __align__(32) bf16_t smPS[4][16 * 32];
  int w = threadIdx.x >> 5;
  bf16_t* PM = &smPM[w][0];
  bf16_t* PS = &smPS[w][0];

  int wid = blockIdx.x * (blockDim.x >> 5) + w;
  int bh = wid >> 6, rb = wid & 63;
  if (bh >= BH) return;
  int r0 = rb << 4;
  int lane = threadIdx.x & 31, c = lane & 15, half8 = (lane >> 4) << 3;
  int b = bh >> 4, h = bh & 15;
  float mix00 = mixp[0], mix01 = mixp[1], mix10 = mixp[2], mix11 = mixp[3];

  v16bf Aq1[2], Aq2[2], Ak1[2], Ak2[2];
#pragma unroll
  for (int ks = 0; ks < 2; ++ks) {
    int off = (bh * NSEQ + r0) * DKH + (ks << 5);
    Aq1[ks] = load_rows(q1 + off, DKH);
    Aq2[ks] = load_rows(q2 + off, DKH);
    Ak1[ks] = load_rows(k1 + off, DKH);
    Ak2[ks] = load_rows(k2 + off, DKH);
  }

  float msS_[8], zSi_[8], ms1_[8], z1i_[8];
#pragma unroll
  for (int i = 0; i < 8; ++i) {
    int idx = bh * NSEQ + r0 + i + half8;
    msS_[i] = msS[idx]; zSi_[i] = 1.f / zS[idx];
    ms1_[i] = ms1[idx]; z1i_[i] = 1.f / z1[idx];
  }

  v8f Ym[4], Uu[4], Al = v8f_zero();
#pragma unroll
  for (int t = 0; t < 4; ++t) { Ym[t] = v8f_zero(); Uu[t] = v8f_zero(); }

  v16bf ones;                                   // B with only column 0 = 1.0
  {
    bf16_t oe = (c == 0) ? (bf16_t)1.0f : (bf16_t)0.0f;
#pragma unroll
    for (int i = 0; i < 16; ++i) ones[i] = oe;
  }

  for (int mb = 0; mb < NSEQ; mb += 32) {
#pragma unroll
    for (int j = 0; j < 2; ++j) {
      int m0 = mb + (j << 4);
      int cb = (bh * NSEQ + m0) * DKH;
      Six S = six_zero();
      accum_six(S, Aq1, Aq2, Ak1, Ak2, k1 + cb, k2 + cb, q1 + cb, q2 + cb);
      int col = m0 + c;
      float anc = anchor[bh * NSEQ + col];
#pragma unroll
      for (int i = 0; i < 8; ++i) {
        int row = r0 + i + half8;
        bool valid = mask[(b * NSEQ + row) * NSEQ + col] != 0;
        float s1v = SCALE * S.p11[i];
        float sm  = SCALE * (mix00 * S.p11[i] + mix01 * S.p12[i] +
                             mix10 * S.p21[i] + mix11 * S.p22[i] +
                             T1C * S.t1[i] + T2C * S.t2[i]);
        float pm = valid ? __expf(sm - msS_[i]) * zSi_[i] : 0.f;
        float ps = valid ? __expf(s1v - ms1_[i]) * z1i_[i] * anc : 0.f;
        int lidx = (i + half8) * 32 + (j << 4) + c;
        PM[lidx] = (bf16_t)pm;
        PS[lidx] = (bf16_t)ps;
      }
    }
    asm volatile("s_wait_dscnt 0" ::: "memory");   // wave-local LDS store->load fence
    v16bf Apm = load_rows(PM, 32);                 // C-layout tile -> A-layout operand
    v16bf Aps = load_rows(PS, 32);
#pragma unroll
    for (int t = 0; t < 4; ++t) {
      v16bf Bv = load_cols(v1t + (bh * DKH + (t << 4)) * NSEQ + mb, NSEQ);
      Ym[t] = wmma_bf16(Apm, Bv, Ym[t]);
      Uu[t] = wmma_bf16(Aps, Bv, Uu[t]);
    }
    Al = wmma_bf16(Aps, ones, Al);                 // alpha accumulates in column 0
  }

#pragma unroll
  for (int i = 0; i < 8; ++i) {
    float a = __shfl(Al[i], lane & 16, 32);        // broadcast col-0 alpha to half-wave
    float inv = 0.5f / (a + 1e-9f);
    int row = r0 + i + half8;
    bf16_t* yrow = Yb + ((b * NSEQ + row) * DIM + h * DKH);
#pragma unroll
    for (int t = 0; t < 4; ++t) {
      float v = 0.5f * Ym[t][i] + Uu[t][i] * inv;  // (1-PRIOR_W)*A_mix + PRIOR_W*A_sharp
      yrow[(t << 4) + c] = (bf16_t)v;
    }
  }
}

// ================= K6: output projection out = Y @ Wp^T (fp32 out) ========
// one wave computes a 16x64 strip; A loaded once per k-step, reused 4x.
__global__ void k_proj(const bf16_t* __restrict__ Yb, const bf16_t* __restrict__ wpb,
                       float* __restrict__ out) {
  int wid = blockIdx.x * (blockDim.x >> 5) + (threadIdx.x >> 5);
  if (wid >= 128 * 16) return;
  int rt = wid >> 4, ct = wid & 15;
  int r0 = rt << 4, o0 = ct << 6;
  const bf16_t* arow = Yb + r0 * DIM;
  const bf16_t* brow = wpb + o0 * DIM;
  v8f acc0 = v8f_zero(), acc1 = v8f_zero(), acc2 = v8f_zero(), acc3 = v8f_zero();
  for (int kk = 0; kk < 32; ++kk) {
    __builtin_prefetch(arow + ((kk + 1) << 5), 0, 3);
    v16bf a  = load_rows(arow + (kk << 5), DIM);
    v16bf b0 = load_cols(brow + (kk << 5), DIM);
    v16bf b1 = load_cols(brow + 16 * DIM + (kk << 5), DIM);
    v16bf b2 = load_cols(brow + 32 * DIM + (kk << 5), DIM);
    v16bf b3 = load_cols(brow + 48 * DIM + (kk << 5), DIM);
    acc0 = wmma_bf16(a, b0, acc0);
    acc1 = wmma_bf16(a, b1, acc1);
    acc2 = wmma_bf16(a, b2, acc2);
    acc3 = wmma_bf16(a, b3, acc3);
  }
  int lane = threadIdx.x & 31, c = lane & 15, half8 = (lane >> 4) << 3;
  v8f accs[4] = {acc0, acc1, acc2, acc3};
#pragma unroll
  for (int cc = 0; cc < 4; ++cc) {
#pragma unroll
    for (int i = 0; i < 8; ++i) {
      int r = r0 + i + half8;
      out[r * DIM + o0 + (cc << 4) + c] = accs[cc][i];
    }
  }
}

// =========================================================================
extern "C" void kernel_launch(void* const* d_in, const int* in_sizes, int n_in,
                              void* d_out, int out_size, void* d_ws, size_t ws_size,
                              hipStream_t stream) {
  (void)in_sizes; (void)n_in; (void)out_size; (void)ws_size;
  const float* x            = (const float*)d_in[0];
  const unsigned char* mask = (const unsigned char*)d_in[1];
  const float* W1           = (const float*)d_in[2];
  const float* W2           = (const float*)d_in[3];
  const float* Wp           = (const float*)d_in[4];
  const float* mix          = (const float*)d_in[5];
  float* out                = (float*)d_out;

  // workspace carve-out (~44 MB), 256B aligned slices
  char* w = (char*)d_ws;
  auto carve = [&](size_t bytes) { char* p = w; w += (bytes + 255) & ~(size_t)255; return p; };
  bf16_t* xb   = (bf16_t*)carve((size_t)2048 * 1024 * 2);
  bf16_t* w1b  = (bf16_t*)carve((size_t)3145728 * 2);
  bf16_t* w2b  = (bf16_t*)carve((size_t)3145728 * 2);
  bf16_t* wpb  = (bf16_t*)carve((size_t)1048576 * 2);
  bf16_t* q1   = (bf16_t*)carve((size_t)2097152 * 2);
  bf16_t* k1   = (bf16_t*)carve((size_t)2097152 * 2);
  bf16_t* q2   = (bf16_t*)carve((size_t)2097152 * 2);
  bf16_t* k2   = (bf16_t*)carve((size_t)2097152 * 2);
  bf16_t* v1t  = (bf16_t*)carve((size_t)2097152 * 2);
  bf16_t* Yb   = (bf16_t*)carve((size_t)2097152 * 2);
  float* msS   = (float*)carve((size_t)BH * NSEQ * 4);
  float* zS    = (float*)carve((size_t)BH * NSEQ * 4);
  float* ms1   = (float*)carve((size_t)BH * NSEQ * 4);
  float* z1    = (float*)carve((size_t)BH * NSEQ * 4);
  float* ms2   = (float*)carve((size_t)BH * NSEQ * 4);
  float* z2    = (float*)carve((size_t)BH * NSEQ * 4);
  float* anch  = (float*)carve((size_t)BH * NSEQ * 4);
  int*   kstar = (int*)carve((size_t)BH * 4);

  k_cvt<<<4096, 256, 0, stream>>>(x,  xb,  2097152);
  k_cvt<<<4096, 256, 0, stream>>>(W1, w1b, 3145728);
  k_cvt<<<4096, 256, 0, stream>>>(W2, w2b, 3145728);
  k_cvt<<<4096, 256, 0, stream>>>(Wp, wpb, 1048576);

  // 2 sets x 128 rowblocks x 48 strips = 12288 waves, 4 waves/block
  k_qkv<<<3072, 128, 0, stream>>>(xb, w1b, w2b, q1, k1, v1t, q2, k2);
  k_attn_stats<<<512, 128, 0, stream>>>(q1, k1, q2, k2, mask, mix,
                                        msS, zS, ms1, z1, ms2, z2);
  k_rowsum_argmax<<<BH, 256, 0, stream>>>(q2, k2, mask, ms2, z2, kstar);
  k_anchor<<<BH, 256, 0, stream>>>(q2, k2, mask, ms2, z2, kstar, anch);
  k_attn_out<<<512, 128, 0, stream>>>(q1, k1, q2, k2, v1t, mask, mix,
                                      msS, zS, ms1, z1, anch, Yb);
  // 128 rowblocks x 16 strips = 2048 waves
  k_proj<<<512, 128, 0, stream>>>(Yb, wpb, out);
}
